// Qwen2MoeSparseMoeBlock_9440338117222
// MI455X (gfx1250) — compile-verified, matrix operand order checked by reading, not run on hardware
//
#include <hip/hip_runtime.h>

// ---------------- problem constants ----------------
#define T_TOK 8192            // B*S
#define H_DIM 2048
#define E_NUM 16
#define I_DIM 1408
#define IS_DIM 5632
#define TOPK 4

// ---------------- GEMM tiling ----------------
#define BM 128
#define BN 64
#define BK 32
#define LDK 40                // padded LDS row stride (bf16 elems): 64B data + 16B pad

typedef __attribute__((ext_vector_type(16))) __bf16 bf16x16;
typedef __attribute__((ext_vector_type(8)))  __bf16 bf16x8;
typedef __attribute__((ext_vector_type(8)))  float  v8f;
typedef __attribute__((ext_vector_type(4)))  unsigned int u32x4;
typedef __attribute__((ext_vector_type(8)))  int i32x8;
typedef __attribute__((ext_vector_type(4)))  int i32x4;

#if defined(__has_builtin)
#if __has_builtin(__builtin_amdgcn_tensor_load_to_lds) && \
    __has_builtin(__builtin_amdgcn_s_wait_tensorcnt)
#define HAVE_TDM 1
#endif
#endif
#ifndef HAVE_TDM
#define HAVE_TDM 0
#endif

__device__ __forceinline__ v8f wmma_bf16(bf16x16 a, bf16x16 b, v8f c) {
  return __builtin_amdgcn_wmma_f32_16x16x32_bf16(false, a, false, b, (short)0, c,
                                                 false, false);
}

// Load one 16x32 bf16 fragment from a padded LDS tile [rows][LDK].
__device__ __forceinline__ bf16x16 ld_frag(const __bf16* tile, int rowBase, int lane) {
  int r    = rowBase + (lane & 15);
  int koff = (lane >> 4) * 8;
  const __bf16* p = tile + r * LDK + koff;
  bf16x8 lo = *(const bf16x8*)(p);
  bf16x8 hi = *(const bf16x8*)(p + 16);
  bf16x16 v;
#pragma unroll
  for (int i = 0; i < 8; ++i) { v[i] = lo[i]; v[i + 8] = hi[i]; }
  return v;
}

__device__ __forceinline__ float fast_silu_mul(float g, float u) {
  return g * __builtin_amdgcn_rcpf(1.f + __expf(-g)) * u;
}

// ---------------------------------------------------------------------
// Tensor Data Mover: async 2D bf16 tile (128 rows x 32 cols) -> LDS with
// hardware padding producing the LDK=40 (80B) row stride.
// OOB rows (>= valid_rows) read zero via tensor_dim1.
// Call from ONE wave only; complete with tdm_wait<N>().
// ---------------------------------------------------------------------
__device__ __forceinline__ void a_tile_load(const __bf16* gsrc, __bf16* ldst,
                                            int valid_rows, int stride_elems,
                                            int lane) {
#if HAVE_TDM
  uint64_t ga = (uint64_t)(uintptr_t)gsrc;
  u32x4 g0;
  g0.x = 1u;                                   // count=1 valid descriptor
  g0.y = (uint32_t)(uintptr_t)ldst;            // lds_addr (low 32b of generic ptr)
  g0.z = (uint32_t)ga;                         // global_addr[31:0]
  g0.w = (uint32_t)((ga >> 32) & 0x1FFFFFFu)   // global_addr[56:32]
         | (2u << 30);                         // type=2 ("image")
  i32x8 g1;
  g1[0] = (1 << 16)                            // data_size=1 -> 2 bytes
        | (1 << 20)                            // pad_enable
        | (3 << 22)                            // pad_interval: every 16 DW (64B)
        | (3 << 25);                           // pad_amount: 4 DW (16B)
  g1[1] = (int)(32u << 16);                    // tensor_dim0 = 32 (cols)
  g1[2] = (int)(((uint32_t)valid_rows & 0xFFFFu) << 16);   // tensor_dim1 lo16
  g1[3] = (int)((((uint32_t)valid_rows >> 16) & 0xFFFFu)   // tensor_dim1 hi16
                | (32u << 16));                // tile_dim0 = 32
  g1[4] = BM;                                  // tile_dim1 = 128, tile_dim2 = 0
  g1[5] = stride_elems;                        // tensor_dim0_stride[31:0]
  g1[6] = 0;
  g1[7] = 0;
  i32x4 z4 = {0, 0, 0, 0};
  i32x8 z8 = {0, 0, 0, 0, 0, 0, 0, 0};
  __builtin_amdgcn_tensor_load_to_lds(g0, g1, z4, z4, z8, 0);
  (void)lane;
#else
  // fallback: the issuing wave copies the tile itself
  for (int c = lane; c < BM * 4; c += 32) {
    int r = c >> 2, q = (c & 3) * 8;
    bf16x8 v = {};
    if (r < valid_rows) v = *(const bf16x8*)(gsrc + (size_t)r * stride_elems + q);
    *(bf16x8*)(ldst + r * LDK + q) = v;
  }
#endif
}

template <int N>
__device__ __forceinline__ void tdm_wait() {
#if HAVE_TDM
  __builtin_amdgcn_s_wait_tensorcnt(N);
#endif
}

// fp32 weights -> bf16 LDS tile row chunk (16 elems), with prefetch
__device__ __forceinline__ void b_chunk_load(const float* src, __bf16* dst) {
  bf16x8 v0, v1;
#pragma unroll
  for (int i = 0; i < 8; ++i) v0[i] = (__bf16)src[i];
#pragma unroll
  for (int i = 0; i < 8; ++i) v1[i] = (__bf16)src[8 + i];
  *(bf16x8*)dst       = v0;
  *(bf16x8*)(dst + 8) = v1;
  __builtin_prefetch(src + BK, 0, 1);
}

// =====================================================================
// 1) Router: logits, softmax, top-4, sigmoid shared gate; emit x in bf16
// =====================================================================
__global__ __launch_bounds__(256) void router_kernel(
    const float* __restrict__ x, const float* __restrict__ rw,
    const float* __restrict__ sew, __bf16* __restrict__ xb,
    int* __restrict__ topk_idx, float* __restrict__ topk_w,
    float* __restrict__ seg) {
  const int t   = blockIdx.x;
  const int tid = threadIdx.x;
  const int lane = tid & 31, w = tid >> 5;
  const float* xr = x + (size_t)t * H_DIM;

  __shared__ float sred[8][17];
  __shared__ float logits[17];

  float xv[8];
#pragma unroll
  for (int i = 0; i < 8; ++i) {
    int h = tid + i * 256;
    xv[i] = xr[h];
    xb[(size_t)t * H_DIM + h] = (__bf16)xv[i];
  }

  float part[17];
#pragma unroll
  for (int e = 0; e < 17; ++e) {
    const float* wr = (e < 16) ? (rw + (size_t)e * H_DIM) : sew;
    float p = 0.f;
#pragma unroll
    for (int i = 0; i < 8; ++i) p += xv[i] * wr[tid + i * 256];
    part[e] = p;
  }
#pragma unroll
  for (int e = 0; e < 17; ++e) {
    float p = part[e];
#pragma unroll
    for (int o = 16; o > 0; o >>= 1) p += __shfl_xor(p, o, 32);
    if (lane == 0) sred[w][e] = p;
  }
  __syncthreads();
  if (tid < 17) {
    float s = 0.f;
#pragma unroll
    for (int ww = 0; ww < 8; ++ww) s += sred[ww][tid];
    logits[tid] = s;
  }
  __syncthreads();
  if (tid == 0) {
    float probs[16];
    float mx = -1e30f;
#pragma unroll
    for (int e = 0; e < 16; ++e) mx = fmaxf(mx, logits[e]);
    float sum = 0.f;
#pragma unroll
    for (int e = 0; e < 16; ++e) { probs[e] = __expf(logits[e] - mx); sum += probs[e]; }
    float inv = __builtin_amdgcn_rcpf(sum);
#pragma unroll
    for (int e = 0; e < 16; ++e) probs[e] *= inv;
#pragma unroll
    for (int k = 0; k < TOPK; ++k) {
      int bi = 0; float bv = probs[0];
#pragma unroll
      for (int e = 1; e < 16; ++e) if (probs[e] > bv) { bv = probs[e]; bi = e; }
      topk_idx[t * TOPK + k] = bi;
      topk_w[t * TOPK + k]   = bv;
      probs[bi] = -1.f;
    }
    seg[t] = __builtin_amdgcn_rcpf(1.f + __expf(-logits[16]));
  }
}

// =====================================================================
// 2) token gather per expert (atomic append; order-independent output)
// =====================================================================
__global__ void zero_counts_kernel(int* counts) {
  if (threadIdx.x < E_NUM) counts[threadIdx.x] = 0;
}

__global__ void gather_kernel(const int* __restrict__ topk_idx,
                              int* __restrict__ counts, int* __restrict__ lists) {
  int t = blockIdx.x * blockDim.x + threadIdx.x;
  if (t >= T_TOK) return;
#pragma unroll
  for (int k = 0; k < TOPK; ++k) {
    int e = topk_idx[t * TOPK + k];
    int pos = atomicAdd(&counts[e], 1);
    lists[e * T_TOK + pos] = t * TOPK + k;   // encode (token, rank)
  }
}

__global__ void prefix_kernel(const int* __restrict__ counts, int* __restrict__ bases) {
  if (threadIdx.x == 0) {
    int s = 0;
    for (int e = 0; e < E_NUM; ++e) { bases[e] = s; s += counts[e]; }
    bases[E_NUM] = s;
  }
}

// =====================================================================
// 3) Expert gate/up GEMM (gathered A rows): H1 = silu(Xg Gw^T)*(Xg Uw^T)
//    grid (T/BM, I/BN, E)
// =====================================================================
__global__ __launch_bounds__(256) void moe_gateup_kernel(
    const __bf16* __restrict__ xb, const float* __restrict__ gate_w,
    const float* __restrict__ up_w, const int* __restrict__ counts,
    const int* __restrict__ bases, const int* __restrict__ lists,
    __bf16* __restrict__ h1) {
  const int e = blockIdx.z;
  const int count = counts[e];
  const int m0 = blockIdx.x * BM;
  if (m0 >= count) return;
  const int n0 = blockIdx.y * BN;

  __shared__ __bf16 sA[BM * LDK];
  __shared__ __bf16 sG[BN * LDK];
  __shared__ __bf16 sU[BN * LDK];
  __shared__ int sEnt[BM];

  const int tid = threadIdx.x;
  if (tid < BM) {
    int m = m0 + tid;
    sEnt[tid] = (m < count) ? lists[e * T_TOK + m] : -1;
  }
  __syncthreads();

  const int lane = tid & 31, w = tid >> 5;
  const int wm = w >> 1, wn = w & 1;

  const int arow = tid >> 1, acol = (tid & 1) * 16;
  const int t2 = tid & 127;
  const int brow = t2 >> 1, bcol = (t2 & 1) * 16;
  const float* bsrc0 = (tid < 128 ? gate_w : up_w) +
                       ((size_t)e * I_DIM + (n0 + brow)) * H_DIM + bcol;
  __bf16* bdst = (tid < 128 ? sG : sU) + brow * LDK + bcol;

  v8f accG[2][2], accU[2][2];
  v8f vz = {0.f, 0.f, 0.f, 0.f, 0.f, 0.f, 0.f, 0.f};
#pragma unroll
  for (int i = 0; i < 2; ++i)
#pragma unroll
    for (int j = 0; j < 2; ++j) { accG[i][j] = vz; accU[i][j] = vz; }

  const int entA = sEnt[arow];
  const __bf16* asrc = (entA >= 0)
      ? (xb + (size_t)(entA >> 2) * H_DIM + acol) : nullptr;

  for (int k0 = 0; k0 < H_DIM; k0 += BK) {
    bf16x8 a0 = {}, a1 = {};
    if (asrc) { a0 = *(const bf16x8*)(asrc + k0); a1 = *(const bf16x8*)(asrc + k0 + 8); }
    *(bf16x8*)(sA + arow * LDK + acol)     = a0;
    *(bf16x8*)(sA + arow * LDK + acol + 8) = a1;
    b_chunk_load(bsrc0 + k0, bdst);
    __syncthreads();

    bf16x16 af0 = ld_frag(sA, wm * 32,      lane);
    bf16x16 af1 = ld_frag(sA, wm * 32 + 16, lane);
    bf16x16 bg0 = ld_frag(sG, wn * 32,      lane);
    bf16x16 bg1 = ld_frag(sG, wn * 32 + 16, lane);
    bf16x16 bu0 = ld_frag(sU, wn * 32,      lane);
    bf16x16 bu1 = ld_frag(sU, wn * 32 + 16, lane);

    accG[0][0] = wmma_bf16(af0, bg0, accG[0][0]);
    accG[0][1] = wmma_bf16(af0, bg1, accG[0][1]);
    accG[1][0] = wmma_bf16(af1, bg0, accG[1][0]);
    accG[1][1] = wmma_bf16(af1, bg1, accG[1][1]);
    accU[0][0] = wmma_bf16(af0, bu0, accU[0][0]);
    accU[0][1] = wmma_bf16(af0, bu1, accU[0][1]);
    accU[1][0] = wmma_bf16(af1, bu0, accU[1][0]);
    accU[1][1] = wmma_bf16(af1, bu1, accU[1][1]);
    __syncthreads();
  }

  const int base = bases[e];
  const int rb = (lane >> 4) * 8;
  const int nlane = lane & 15;
#pragma unroll
  for (int fm = 0; fm < 2; ++fm)
#pragma unroll
    for (int fn = 0; fn < 2; ++fn)
#pragma unroll
      for (int i = 0; i < 8; ++i) {
        int ml = wm * 32 + fm * 16 + rb + i;
        int mg = m0 + ml;
        if (mg < count) {
          float h = fast_silu_mul(accG[fm][fn][i], accU[fm][fn][i]);
          int nc = n0 + wn * 32 + fn * 16 + nlane;
          h1[(size_t)(base + mg) * I_DIM + nc] = (__bf16)h;
        }
      }
}

// =====================================================================
// 4) Expert down GEMM (dense compacted A via TDM, double-buffered)
//    -> weighted slot (t, rank): slots[ent*H + h].  grid (T/BM, H/BN, E)
// =====================================================================
__global__ __launch_bounds__(256) void moe_down_kernel(
    const __bf16* __restrict__ h1, const float* __restrict__ down_w,
    const int* __restrict__ counts, const int* __restrict__ bases,
    const int* __restrict__ lists, const float* __restrict__ topk_w,
    float* __restrict__ slots) {
  const int e = blockIdx.z;
  const int count = counts[e];
  const int m0 = blockIdx.x * BM;
  if (m0 >= count) return;
  const int n0 = blockIdx.y * BN;
  const int base = bases[e];

  __shared__ __bf16 sA[2][BM * LDK];
  __shared__ __bf16 sB[2][BN * LDK];
  __shared__ int sEnt[BM];

  const int tid = threadIdx.x;
  if (tid < BM) {
    int m = m0 + tid;
    sEnt[tid] = (m < count) ? lists[e * T_TOK + m] : -1;
  }

  const int lane = tid & 31, w = tid >> 5;
  const int wm = w >> 1, wn = w & 1;
  const int brow = (tid & 127) >> 1, bcol = ((tid & 127) & 1) * 16;

  const __bf16* abase = h1 + (size_t)(base + m0) * I_DIM;
  const int vrows = count - m0;                       // >= 1; rows past -> zero
  const float* bsrc0 = down_w + ((size_t)e * H_DIM + (n0 + brow)) * I_DIM + bcol;
  __bf16* bdst0 = &sB[0][brow * LDK + bcol];
  __bf16* bdst1 = &sB[1][brow * LDK + bcol];

  v8f acc[2][2];
  v8f vz = {0.f, 0.f, 0.f, 0.f, 0.f, 0.f, 0.f, 0.f};
#pragma unroll
  for (int i = 0; i < 2; ++i)
#pragma unroll
    for (int j = 0; j < 2; ++j) acc[i][j] = vz;

  const int ITERS = I_DIM / BK;                       // 44
  if (tid < 32) a_tile_load(abase, &sA[0][0], vrows, I_DIM, lane);
  if (tid < 128) b_chunk_load(bsrc0, bdst0);

  for (int i = 0; i < ITERS; ++i) {
    const int cur = i & 1;
    const int k1 = (i + 1) * BK;
    if (i + 1 < ITERS) {
      if (tid < 32) a_tile_load(abase + k1, &sA[cur ^ 1][0], vrows, I_DIM, lane);
      if (tid < 128) b_chunk_load(bsrc0 + k1, cur ? bdst0 : bdst1);
      if (tid < 32) tdm_wait<1>();
    } else {
      if (tid < 32) tdm_wait<0>();
    }
    __syncthreads();

    bf16x16 af0 = ld_frag(&sA[cur][0], wm * 32,      lane);
    bf16x16 af1 = ld_frag(&sA[cur][0], wm * 32 + 16, lane);
    bf16x16 bf0 = ld_frag(&sB[cur][0], wn * 32,      lane);
    bf16x16 bf1 = ld_frag(&sB[cur][0], wn * 32 + 16, lane);

    acc[0][0] = wmma_bf16(af0, bf0, acc[0][0]);
    acc[0][1] = wmma_bf16(af0, bf1, acc[0][1]);
    acc[1][0] = wmma_bf16(af1, bf0, acc[1][0]);
    acc[1][1] = wmma_bf16(af1, bf1, acc[1][1]);
    __syncthreads();
  }

  const int rb = (lane >> 4) * 8;
  const int nlane = lane & 15;
#pragma unroll
  for (int fm = 0; fm < 2; ++fm)
#pragma unroll
    for (int fn = 0; fn < 2; ++fn)
#pragma unroll
      for (int i = 0; i < 8; ++i) {
        int ml = wm * 32 + fm * 16 + rb + i;
        int ent = sEnt[ml];
        if (ent >= 0) {
          float wgt = topk_w[ent];
          int nc = n0 + wn * 32 + fn * 16 + nlane;
          slots[(size_t)ent * H_DIM + nc] = wgt * acc[fm][fn][i];
        }
      }
}

// =====================================================================
// 5) Shared expert gate/up (dense A via TDM, double-buffered)
//    grid (T/BM, IS/BN)
// =====================================================================
__global__ __launch_bounds__(256) void shared_gateup_kernel(
    const __bf16* __restrict__ xb, const float* __restrict__ sgw,
    const float* __restrict__ suw, __bf16* __restrict__ hs) {
  const int m0 = blockIdx.x * BM;
  const int n0 = blockIdx.y * BN;

  __shared__ __bf16 sA[2][BM * LDK];
  __shared__ __bf16 sG[2][BN * LDK];
  __shared__ __bf16 sU[2][BN * LDK];

  const int tid = threadIdx.x;
  const int lane = tid & 31, w = tid >> 5;
  const int wm = w >> 1, wn = w & 1;
  const int t2 = tid & 127;
  const int brow = t2 >> 1, bcol = (t2 & 1) * 16;

  const __bf16* abase = xb + (size_t)m0 * H_DIM;
  const float* bsrc0 = (tid < 128 ? sgw : suw) + (size_t)(n0 + brow) * H_DIM + bcol;
  __bf16* bdst0 = (tid < 128 ? &sG[0][0] : &sU[0][0]) + brow * LDK + bcol;
  __bf16* bdst1 = (tid < 128 ? &sG[1][0] : &sU[1][0]) + brow * LDK + bcol;

  v8f accG[2][2], accU[2][2];
  v8f vz = {0.f, 0.f, 0.f, 0.f, 0.f, 0.f, 0.f, 0.f};
#pragma unroll
  for (int i = 0; i < 2; ++i)
#pragma unroll
    for (int j = 0; j < 2; ++j) { accG[i][j] = vz; accU[i][j] = vz; }

  const int ITERS = H_DIM / BK;                       // 64
  if (tid < 32) a_tile_load(abase, &sA[0][0], BM, H_DIM, lane);
  b_chunk_load(bsrc0, bdst0);

  for (int i = 0; i < ITERS; ++i) {
    const int cur = i & 1;
    const int k1 = (i + 1) * BK;
    if (i + 1 < ITERS) {
      if (tid < 32) a_tile_load(abase + k1, &sA[cur ^ 1][0], BM, H_DIM, lane);
      b_chunk_load(bsrc0 + k1, cur ? bdst0 : bdst1);
      if (tid < 32) tdm_wait<1>();
    } else {
      if (tid < 32) tdm_wait<0>();
    }
    __syncthreads();

    bf16x16 af0 = ld_frag(&sA[cur][0], wm * 32,      lane);
    bf16x16 af1 = ld_frag(&sA[cur][0], wm * 32 + 16, lane);
    bf16x16 bg0 = ld_frag(&sG[cur][0], wn * 32,      lane);
    bf16x16 bg1 = ld_frag(&sG[cur][0], wn * 32 + 16, lane);
    bf16x16 bu0 = ld_frag(&sU[cur][0], wn * 32,      lane);
    bf16x16 bu1 = ld_frag(&sU[cur][0], wn * 32 + 16, lane);

    accG[0][0] = wmma_bf16(af0, bg0, accG[0][0]);
    accG[0][1] = wmma_bf16(af0, bg1, accG[0][1]);
    accG[1][0] = wmma_bf16(af1, bg0, accG[1][0]);
    accG[1][1] = wmma_bf16(af1, bg1, accG[1][1]);
    accU[0][0] = wmma_bf16(af0, bu0, accU[0][0]);
    accU[0][1] = wmma_bf16(af0, bu1, accU[0][1]);
    accU[1][0] = wmma_bf16(af1, bu0, accU[1][0]);
    accU[1][1] = wmma_bf16(af1, bu1, accU[1][1]);
    __syncthreads();
  }

  const int rb = (lane >> 4) * 8;
  const int nlane = lane & 15;
#pragma unroll
  for (int fm = 0; fm < 2; ++fm)
#pragma unroll
    for (int fn = 0; fn < 2; ++fn)
#pragma unroll
      for (int i = 0; i < 8; ++i) {
        int m = m0 + wm * 32 + fm * 16 + rb + i;
        float h = fast_silu_mul(accG[fm][fn][i], accU[fm][fn][i]);
        int nc = n0 + wn * 32 + fn * 16 + nlane;
        hs[(size_t)m * IS_DIM + nc] = (__bf16)h;
      }
}

// =====================================================================
// 6) Shared down GEMM (dense A via TDM) fused with final combine:
//    out = sigmoid(se)*shared_down + sum_k slots[t,k].  grid (T/BM, H/BN)
// =====================================================================
__global__ __launch_bounds__(256) void shared_down_combine_kernel(
    const __bf16* __restrict__ hs, const float* __restrict__ sdw,
    const float* __restrict__ seg, const float* __restrict__ slots,
    float* __restrict__ out) {
  const int m0 = blockIdx.x * BM;
  const int n0 = blockIdx.y * BN;

  __shared__ __bf16 sA[2][BM * LDK];
  __shared__ __bf16 sB[2][BN * LDK];

  const int tid = threadIdx.x;
  const int lane = tid & 31, w = tid >> 5;
  const int wm = w >> 1, wn = w & 1;
  const int brow = (tid & 127) >> 1, bcol = ((tid & 127) & 1) * 16;

  const __bf16* abase = hs + (size_t)m0 * IS_DIM;
  const float* bsrc0 = sdw + (size_t)(n0 + brow) * IS_DIM + bcol;
  __bf16* bdst0 = &sB[0][brow * LDK + bcol];
  __bf16* bdst1 = &sB[1][brow * LDK + bcol];

  v8f acc[2][2];
  v8f vz = {0.f, 0.f, 0.f, 0.f, 0.f, 0.f, 0.f, 0.f};
#pragma unroll
  for (int i = 0; i < 2; ++i)
#pragma unroll
    for (int j = 0; j < 2; ++j) acc[i][j] = vz;

  const int ITERS = IS_DIM / BK;                      // 176
  if (tid < 32) a_tile_load(abase, &sA[0][0], BM, IS_DIM, lane);
  if (tid < 128) b_chunk_load(bsrc0, bdst0);

  for (int i = 0; i < ITERS; ++i) {
    const int cur = i & 1;
    const int k1 = (i + 1) * BK;
    if (i + 1 < ITERS) {
      if (tid < 32) a_tile_load(abase + k1, &sA[cur ^ 1][0], BM, IS_DIM, lane);
      if (tid < 128) b_chunk_load(bsrc0 + k1, cur ? bdst0 : bdst1);
      if (tid < 32) tdm_wait<1>();
    } else {
      if (tid < 32) tdm_wait<0>();
    }
    __syncthreads();

    bf16x16 af0 = ld_frag(&sA[cur][0], wm * 32,      lane);
    bf16x16 af1 = ld_frag(&sA[cur][0], wm * 32 + 16, lane);
    bf16x16 bf0 = ld_frag(&sB[cur][0], wn * 32,      lane);
    bf16x16 bf1 = ld_frag(&sB[cur][0], wn * 32 + 16, lane);

    acc[0][0] = wmma_bf16(af0, bf0, acc[0][0]);
    acc[0][1] = wmma_bf16(af0, bf1, acc[0][1]);
    acc[1][0] = wmma_bf16(af1, bf0, acc[1][0]);
    acc[1][1] = wmma_bf16(af1, bf1, acc[1][1]);
    __syncthreads();
  }

  const int rb = (lane >> 4) * 8;
  const int nlane = lane & 15;
#pragma unroll
  for (int fm = 0; fm < 2; ++fm)
#pragma unroll
    for (int fn = 0; fn < 2; ++fn)
#pragma unroll
      for (int i = 0; i < 8; ++i) {
        int t = m0 + wm * 32 + fm * 16 + rb + i;
        int nc = n0 + wn * 32 + fn * 16 + nlane;
        float r = seg[t] * acc[fm][fn][i];
        const float* sl = slots + (size_t)t * TOPK * H_DIM + nc;
        r += sl[0] + sl[H_DIM] + sl[2 * H_DIM] + sl[3 * H_DIM];
        out[(size_t)t * H_DIM + nc] = r;
      }
}

// =====================================================================
// launch
// =====================================================================
extern "C" void kernel_launch(void* const* d_in, const int* in_sizes, int n_in,
                              void* d_out, int out_size, void* d_ws, size_t ws_size,
                              hipStream_t stream) {
  (void)in_sizes; (void)n_in; (void)out_size; (void)ws_size;
  const float* x   = (const float*)d_in[0];
  const float* rw  = (const float*)d_in[1];
  const float* gw  = (const float*)d_in[2];
  const float* uw  = (const float*)d_in[3];
  const float* dw  = (const float*)d_in[4];
  const float* sgw = (const float*)d_in[5];
  const float* suw = (const float*)d_in[6];
  const float* sdw = (const float*)d_in[7];
  const float* sew = (const float*)d_in[8];
  float* out = (float*)d_out;

  char* ws = (char*)d_ws;
  size_t off = 0;
  auto alloc = [&](size_t bytes) -> void* {
    void* p = ws + off;
    off = (off + bytes + 255) & ~(size_t)255;
    return p;
  };

  __bf16* xb    = (__bf16*)alloc((size_t)T_TOK * H_DIM * 2);
  int*    tki   = (int*)   alloc((size_t)T_TOK * TOPK * 4);
  float*  tkw   = (float*) alloc((size_t)T_TOK * TOPK * 4);
  float*  seg   = (float*) alloc((size_t)T_TOK * 4);
  int*    cnts  = (int*)   alloc((size_t)E_NUM * 4);
  int*    bases = (int*)   alloc((size_t)(E_NUM + 1) * 4);
  int*    lists = (int*)   alloc((size_t)E_NUM * T_TOK * 4);
  __bf16* h1    = (__bf16*)alloc((size_t)T_TOK * TOPK * I_DIM * 2);
  __bf16* hs    = (__bf16*)alloc((size_t)T_TOK * IS_DIM * 2);
  float*  slots = (float*) alloc((size_t)T_TOK * TOPK * H_DIM * 4);

  router_kernel<<<T_TOK, 256, 0, stream>>>(x, rw, sew, xb, tki, tkw, seg);
  zero_counts_kernel<<<1, 32, 0, stream>>>(cnts);
  gather_kernel<<<T_TOK / 256, 256, 0, stream>>>(tki, cnts, lists);
  prefix_kernel<<<1, 32, 0, stream>>>(cnts, bases);

  moe_gateup_kernel<<<dim3(T_TOK / BM, I_DIM / BN, E_NUM), 256, 0, stream>>>(
      xb, gw, uw, cnts, bases, lists, h1);
  moe_down_kernel<<<dim3(T_TOK / BM, H_DIM / BN, E_NUM), 256, 0, stream>>>(
      h1, dw, cnts, bases, lists, tkw, slots);

  shared_gateup_kernel<<<dim3(T_TOK / BM, IS_DIM / BN), 256, 0, stream>>>(
      xb, sgw, suw, hs);
  shared_down_combine_kernel<<<dim3(T_TOK / BM, H_DIM / BN), 256, 0, stream>>>(
      hs, sdw, seg, slots, out);
}
